// Encoder_47210280517649
// MI455X (gfx1250) — compile-verified, matrix operand order checked by reading, not run on hardware
//
#include <hip/hip_runtime.h>
#include <hip/hip_bf16.h>

// ---------------------------------------------------------------------------
// Pyraformer-style encoder forward for gfx1250.
// All GEMMs: f16 operands (pre-converted), f32 WMMA accumulation,
// v_wmma_f32_16x16x32_f16, 2x4 tiles (32x64) per wave, 4 waves/block.
// Branch-free inner loop: row/col indices clamped (memory-safe, unsaved rows
// absorb garbage), K-tail hoisted out of the loop (only reduction dim needs
// zero-fill).  Attention sparsity via per-q-tile key bitmask in SGPRs.
// ---------------------------------------------------------------------------

#define NBATCH  16
#define SEQLEN  512
#define DMODEL  512
#define LTOT    680
#define NHEAD   8
#define DKV     64
#define DFF     2048
#define DBN     128
#define NLAYER  4
#define NTILE_L 43                 // ceil(680/16)

typedef __attribute__((ext_vector_type(16))) _Float16 v16h;
typedef __attribute__((ext_vector_type(8)))  _Float16 v8h;
typedef __attribute__((ext_vector_type(8)))  float    v8f;

// ---- f16 fragment loaders --------------------------------------------------
// Fragment layout: lane kh=lane>>4, m/n=lane&15; element e holds
// k = k0 + kh*8 + (e<8 ? e : e+8).  p = row base pointer (pre-clamped).
__device__ __forceinline__ v16h frag_fast(const _Float16* __restrict__ p,
                                          int k0, int kh)
{
    v16h f;
    v8h lo = *(const v8h*)(p + k0 + kh * 8);
    v8h hi = *(const v8h*)(p + k0 + kh * 8 + 16);
#pragma unroll
    for (int e = 0; e < 8; ++e) { f[e] = lo[e]; f[8 + e] = hi[e]; }
    return f;
}

__device__ __forceinline__ v16h frag_tail(const _Float16* __restrict__ p,
                                          int k0, int kh, int K)
{
    v16h f;
#pragma unroll
    for (int e = 0; e < 16; ++e) {
        int k = k0 + kh * 8 + (e < 8 ? e : e + 8);
        f[e] = (k < K) ? p[k] : (_Float16)0.0f;
    }
    return f;
}

// ---- batched WMMA GEMM: C = A[M,K] * Bt[N,K]^T (+bias) ---------------------
// TC: float or _Float16 output.  STORE_T: store C transposed (C[n*ldc+m]).
// SKIP: 0 = none, 1 = skip output group when bitmask bits [nt0..nt0+3] clear
// (scores GEMM; stale tiles re-masked in softmax), 2 = skip K-block when
// bitmask bits [kt0..kt0+1] clear (attn@V; those probabilities are exactly 0).
template <typename TC, bool STORE_T, int SKIP>
__global__ __launch_bounds__(128) void gemm_wmma(
    const _Float16* __restrict__ A, const _Float16* __restrict__ Bt,
    TC* __restrict__ C, const float* __restrict__ bias,
    int M, int N, int K, int lda, int ldb, int ldc,
    long sAb, long sAh, long sBb, long sBh, long sCb, long sCh, int H,
    const unsigned long long* __restrict__ tmask)
{
    int z  = blockIdx.z;
    int zb = z / H;
    int zh = z - zb * H;
    A  += (long)zb * sAb + (long)zh * sAh;
    Bt += (long)zb * sBb + (long)zh * sBh;
    C  += (long)zb * sCb + (long)zh * sCh;

    int wave = threadIdx.x >> 5;
    int lane = threadIdx.x & 31;
    int kh   = lane >> 4;
    int mt0  = blockIdx.x * 8 + wave * 2;   // 2 M-tiles per wave
    int nt0  = blockIdx.y * 4;              // 4 N-tiles per wave
    if (mt0 * 16 >= M) return;              // wave-uniform

    unsigned long long rm = 0;
    if (SKIP != 0) {                        // active-key-tile mask -> SGPRs
        unsigned long long r = tmask[mt0];
        if ((mt0 + 1) * 16 < M) r |= tmask[mt0 + 1];
        unsigned int lo = (unsigned int)__builtin_amdgcn_readfirstlane((int)(unsigned int)r);
        unsigned int hi = (unsigned int)__builtin_amdgcn_readfirstlane((int)(unsigned int)(r >> 32));
        rm = ((unsigned long long)hi << 32) | lo;
    }
    if (SKIP == 1) {
        if (((rm >> nt0) & 15ull) == 0) return;
    }

    int mrow0 = mt0 * 16 + (lane & 15);
    int mrow1 = mrow0 + 16;
    // clamp: OOB rows/cols load valid memory; their outputs are never stored
    const _Float16* pa0 = A + (long)(mrow0 < M ? mrow0 : M - 1) * lda;
    const _Float16* pa1 = A + (long)(mrow1 < M ? mrow1 : M - 1) * lda;
    const _Float16* pb[4];
#pragma unroll
    for (int j = 0; j < 4; ++j) {
        int ncol = (nt0 + j) * 16 + (lane & 15);
        pb[j] = Bt + (long)(ncol < N ? ncol : N - 1) * ldb;
    }

    v8f acc[2][4];
#pragma unroll
    for (int i = 0; i < 2; ++i)
#pragma unroll
        for (int j = 0; j < 4; ++j)
#pragma unroll
            for (int r = 0; r < 8; ++r) acc[i][j][r] = 0.f;

    int Kfull = K & ~31;
    for (int k0 = 0; k0 < Kfull; k0 += 32) {
        if (SKIP == 2) {                    // pure scalar bit test, s_cbranch
            if (((rm >> (k0 >> 4)) & 3ull) == 0) continue;
        }
        v16h a0 = frag_fast(pa0, k0, kh);
        v16h a1 = frag_fast(pa1, k0, kh);
#pragma unroll
        for (int j = 0; j < 4; ++j) {
            v16h bf = frag_fast(pb[j], k0, kh);
            acc[0][j] = __builtin_amdgcn_wmma_f32_16x16x32_f16(
                false, a0, false, bf, (short)0, acc[0][j], false, false);
            acc[1][j] = __builtin_amdgcn_wmma_f32_16x16x32_f16(
                false, a1, false, bf, (short)0, acc[1][j], false, false);
        }
    }
    if (Kfull < K) {                        // reduction tail, executed once
        bool go = true;
        if (SKIP == 2) go = ((rm >> (Kfull >> 4)) & 3ull) != 0;
        if (go) {
            v16h a0 = frag_tail(pa0, Kfull, kh, K);
            v16h a1 = frag_tail(pa1, Kfull, kh, K);
#pragma unroll
            for (int j = 0; j < 4; ++j) {
                v16h bf = frag_tail(pb[j], Kfull, kh, K);
                acc[0][j] = __builtin_amdgcn_wmma_f32_16x16x32_f16(
                    false, a0, false, bf, (short)0, acc[0][j], false, false);
                acc[1][j] = __builtin_amdgcn_wmma_f32_16x16x32_f16(
                    false, a1, false, bf, (short)0, acc[1][j], false, false);
            }
        }
    }

#pragma unroll
    for (int j = 0; j < 4; ++j) {
        int ncol = (nt0 + j) * 16 + (lane & 15);
        if (ncol >= N) continue;
        float bvl = bias ? bias[ncol] : 0.f;
#pragma unroll
        for (int i = 0; i < 2; ++i)
#pragma unroll
            for (int r = 0; r < 8; ++r) {
                int m = (mt0 + i) * 16 + kh * 8 + r;   // D: M = r + 8*(lane>>4)
                if (m < M) {
                    float v = acc[i][j][r] + bvl;
                    if (STORE_T) C[(long)ncol * ldc + m] = (TC)v;
                    else         C[(long)m * ldc + ncol] = (TC)v;
                }
            }
    }
}

// ---- weight transpose+convert: f32 [K,N] -> f16 [N,K] ----------------------
__global__ __launch_bounds__(256) void wtrans_kernel(
    const float* __restrict__ in, _Float16* __restrict__ out, int K, int N)
{
    long i = (long)blockIdx.x * blockDim.x + threadIdx.x;
    if (i >= (long)K * N) return;
    int k = (int)(i % K);
    long n = i / K;
    out[n * K + k] = (_Float16)in[(long)k * N + n];
}

// ---- per-q-tile bitmask of key tiles containing any attended entry ---------
__global__ __launch_bounds__(64) void tilemap_kernel(
    const unsigned char* __restrict__ mask, unsigned long long* __restrict__ rmask)
{
    int qt = blockIdx.x * blockDim.x + threadIdx.x;
    if (qt >= NTILE_L) return;
    unsigned long long bits = 0;
    for (int kt = 0; kt < NTILE_L; ++kt) {
        int any = 0;
        for (int i = 0; i < 16; ++i) {
            int q = qt * 16 + i;
            if (q >= LTOT) break;
            for (int j = 0; j < 16; ++j) {
                int k = kt * 16 + j;
                if (k >= LTOT) break;
                if (!mask[(long)q * LTOT + k]) any = 1;
            }
        }
        if (any) bits |= (1ull << kt);
    }
    rmask[qt] = bits;
}

// ---- embedding: circular conv(k=3) + positional + covariate proj ----------
__global__ __launch_bounds__(256) void embed_kernel(
    const float* __restrict__ x_enc, const float* __restrict__ x_mark,
    const float* __restrict__ tokK, const float* __restrict__ markW,
    const float* __restrict__ markb, float* __restrict__ X,
    _Float16* __restrict__ Xh)
{
    long gid = (long)blockIdx.x * blockDim.x + threadIdx.x;
    if (gid >= (long)NBATCH * SEQLEN * DMODEL) return;
    int d  = (int)(gid & (DMODEL - 1));
    long bt = gid >> 9;
    int t  = (int)(bt & (SEQLEN - 1));
    int b  = (int)(bt >> 9);

    float acc = 0.f;
#pragma unroll
    for (int j = 0; j < 3; ++j) {
        int tt = (t - 1 + j + SEQLEN) & (SEQLEN - 1);     // wrap pad
        const float* xe = x_enc + ((long)b * SEQLEN + tt) * 7;
        const float* kk = tokK + (long)(j * 7) * DMODEL + d;
#pragma unroll
        for (int c = 0; c < 7; ++c) acc += xe[c] * kk[(long)c * DMODEL];
    }
    const float* xm = x_mark + ((long)b * SEQLEN + t) * 4;
#pragma unroll
    for (int c = 0; c < 4; ++c) acc += xm[c] * markW[c * DMODEL + d];
    acc += markb[d];

    int d2 = d & ~1;
    float div = expf(-(float)d2 * (logf(10000.f) / (float)DMODEL));
    float ang = (float)t * div;
    acc += (d & 1) ? cosf(ang) : sinf(ang);

    long o = ((long)b * LTOT + t) * DMODEL + d;
    X[o] = acc;
    Xh[o] = (_Float16)acc;
}

// ---- batchnorm stats over (batch, length) per channel ----------------------
__global__ __launch_bounds__(256) void bn_stats_kernel(
    const float* __restrict__ t, int NE, float* __restrict__ stats)
{
    int c = blockIdx.x, tid = threadIdx.x;
    float s = 0.f, s2 = 0.f;
    for (int i = tid; i < NE; i += 256) {
        float v = t[(long)i * DBN + c];
        s += v; s2 += v * v;
    }
    __shared__ float sh0[256], sh1[256];
    sh0[tid] = s; sh1[tid] = s2; __syncthreads();
    for (int off = 128; off > 0; off >>= 1) {
        if (tid < off) { sh0[tid] += sh0[tid + off]; sh1[tid] += sh1[tid + off]; }
        __syncthreads();
    }
    if (tid == 0) {
        float mu = sh0[0] / (float)NE;
        stats[c] = mu;
        stats[DBN + c] = sh1[0] / (float)NE - mu * mu;
    }
}

__global__ __launch_bounds__(256) void bn_elu_kernel(
    const float* __restrict__ in, _Float16* __restrict__ outh,
    const float* __restrict__ stats, const float* __restrict__ g,
    const float* __restrict__ bb, long n)
{
    long i = (long)blockIdx.x * blockDim.x + threadIdx.x;
    if (i >= n) return;
    int c = (int)(i & (DBN - 1));
    float v = (in[i] - stats[c]) * rsqrtf(stats[DBN + c] + 1e-5f) * g[c] + bb[c];
    outh[i] = (_Float16)(v > 0.f ? v : expm1f(v));
}

// ---- LayerNorm(dim 512), optional residual; writes f32 + f16 mirror --------
__global__ __launch_bounds__(256) void ln_kernel(
    const float* __restrict__ a, const float* __restrict__ r,
    float* __restrict__ out, _Float16* __restrict__ outh,
    const float* __restrict__ g, const float* __restrict__ b)
{
    long row = blockIdx.x;
    int tid = threadIdx.x;
    const float* ar = a + row * DMODEL;
    const float* rr = r ? r + row * DMODEL : nullptr;
    float v0 = ar[tid]       + (rr ? rr[tid]       : 0.f);
    float v1 = ar[tid + 256] + (rr ? rr[tid + 256] : 0.f);

    __shared__ float sh[256];
    sh[tid] = v0 + v1; __syncthreads();
    for (int off = 128; off > 0; off >>= 1) {
        if (tid < off) sh[tid] += sh[tid + off];
        __syncthreads();
    }
    float mean = sh[0] * (1.f / DMODEL); __syncthreads();
    float d0 = v0 - mean, d1 = v1 - mean;
    sh[tid] = d0 * d0 + d1 * d1; __syncthreads();
    for (int off = 128; off > 0; off >>= 1) {
        if (tid < off) sh[tid] += sh[tid + off];
        __syncthreads();
    }
    float inv = rsqrtf(sh[0] * (1.f / DMODEL) + 1e-5f);
    float y0 = d0 * inv * g[tid]       + b[tid];
    float y1 = d1 * inv * g[tid + 256] + b[tid + 256];
    out[row * DMODEL + tid]        = y0;
    out[row * DMODEL + tid + 256]  = y1;
    outh[row * DMODEL + tid]       = (_Float16)y0;
    outh[row * DMODEL + tid + 256] = (_Float16)y1;
}

// ---- masked scaled softmax; f32 scores -> f16 probabilities ----------------
__global__ __launch_bounds__(256) void softmax_kernel(
    const float* __restrict__ scores, _Float16* __restrict__ probs,
    const unsigned char* __restrict__ mask)
{
    int q = blockIdx.x, h = blockIdx.y, tid = threadIdx.x;
    const float* row = scores + ((long)h * LTOT + q) * LTOT;
    _Float16* orow = probs + ((long)h * LTOT + q) * LTOT;
    const unsigned char* mrow = mask + (long)q * LTOT;
    __shared__ float sh[256];

    float vloc[3];
    float mx = -3.0e38f;
#pragma unroll
    for (int c = 0; c < 3; ++c) {
        int i = tid + c * 256;
        float v = -3.0e38f;
        if (i < LTOT) {
            v = row[i] * 0.125f;                 // / sqrt(64)
            if (mrow[i]) v = -1.0e9f;            // masked
        }
        vloc[c] = v;
        mx = fmaxf(mx, v);
    }
    sh[tid] = mx; __syncthreads();
    for (int off = 128; off > 0; off >>= 1) {
        if (tid < off) sh[tid] = fmaxf(sh[tid], sh[tid + off]);
        __syncthreads();
    }
    mx = sh[0]; __syncthreads();

    float s = 0.f;
#pragma unroll
    for (int c = 0; c < 3; ++c) {
        int i = tid + c * 256;
        if (i < LTOT) {
            float e = __expf(vloc[c] - mx);      // masked -> exactly 0.0f
            vloc[c] = e; s += e;
        }
    }
    sh[tid] = s; __syncthreads();
    for (int off = 128; off > 0; off >>= 1) {
        if (tid < off) sh[tid] += sh[tid + off];
        __syncthreads();
    }
    float inv = 1.f / sh[0];
#pragma unroll
    for (int c = 0; c < 3; ++c) {
        int i = tid + c * 256;
        if (i < LTOT) orow[i] = (_Float16)(vloc[c] * inv);
    }
}

// ---- exact GELU: f32 in -> f16 out -----------------------------------------
__global__ __launch_bounds__(256) void gelu_kernel(
    const float* __restrict__ x, _Float16* __restrict__ out, long n)
{
    long i = (long)blockIdx.x * blockDim.x + threadIdx.x;
    if (i >= n) return;
    float v = x[i];
    out[i] = (_Float16)(0.5f * v * (1.f + erff(v * 0.70710678118654752f)));
}

// ---- final gather ----------------------------------------------------------
__global__ __launch_bounds__(256) void gather_kernel(
    const float* __restrict__ X, const int* __restrict__ idx,
    float* __restrict__ out)
{
    long i = (long)blockIdx.x * blockDim.x + threadIdx.x;
    if (i >= (long)NBATCH * 2048 * DMODEL) return;
    int d = (int)(i & (DMODEL - 1));
    long bi = i >> 9;
    int g = (int)(bi & 2047);
    int b = (int)(bi >> 11);
    out[i] = X[((long)b * LTOT + idx[g]) * DMODEL + d];
}

// ---------------------------------------------------------------------------
enum { G_F32 = 0, G_F32_MN, G_F16, G_F16_MK, G_F16T };

static inline void launch_gemm(int mode,
    const _Float16* A, const _Float16* Bt, void* C, const float* bias,
    int M, int N, int K, int lda, int ldb, int ldc,
    long sAb, long sAh, long sBb, long sBh, long sCb, long sCh,
    int nb, int H, const unsigned long long* tmask, hipStream_t s)
{
    dim3 grid((M + 127) / 128, (N + 63) / 64, nb * H);
    dim3 block(128, 1, 1);
    switch (mode) {
    case G_F32:
        gemm_wmma<float, false, 0><<<grid, block, 0, s>>>(A, Bt, (float*)C, bias,
            M, N, K, lda, ldb, ldc, sAb, sAh, sBb, sBh, sCb, sCh, H, tmask);
        break;
    case G_F32_MN:
        gemm_wmma<float, false, 1><<<grid, block, 0, s>>>(A, Bt, (float*)C, bias,
            M, N, K, lda, ldb, ldc, sAb, sAh, sBb, sBh, sCb, sCh, H, tmask);
        break;
    case G_F16:
        gemm_wmma<_Float16, false, 0><<<grid, block, 0, s>>>(A, Bt, (_Float16*)C, bias,
            M, N, K, lda, ldb, ldc, sAb, sAh, sBb, sBh, sCb, sCh, H, tmask);
        break;
    case G_F16_MK:
        gemm_wmma<_Float16, false, 2><<<grid, block, 0, s>>>(A, Bt, (_Float16*)C, bias,
            M, N, K, lda, ldb, ldc, sAb, sAh, sBb, sBh, sCb, sCh, H, tmask);
        break;
    default:
        gemm_wmma<_Float16, true, 0><<<grid, block, 0, s>>>(A, Bt, (_Float16*)C, bias,
            M, N, K, lda, ldb, ldc, sAb, sAh, sBb, sBh, sCb, sCh, H, tmask);
        break;
    }
}

extern "C" void kernel_launch(void* const* d_in, const int* in_sizes, int n_in,
                              void* d_out, int out_size, void* d_ws, size_t ws_size,
                              hipStream_t stream)
{
    const float* x_enc  = (const float*)d_in[0];
    const float* x_mark = (const float*)d_in[1];
    const unsigned char* mask = (const unsigned char*)d_in[2];
    const int*   gidx   = (const int*)d_in[3];
    const float* tokK   = (const float*)d_in[4];
    const float* markW  = (const float*)d_in[5];
    const float* markb  = (const float*)d_in[6];
    const float* downW  = (const float*)d_in[7];
    const float* downb  = (const float*)d_in[8];
    const float* convK  = (const float*)d_in[9];
    const float* convb  = (const float*)d_in[10];
    const float* bng    = (const float*)d_in[11];
    const float* bnb    = (const float*)d_in[12];
    const float* upW    = (const float*)d_in[13];
    const float* upb    = (const float*)d_in[14];
    const float* cng    = (const float*)d_in[15];
    const float* cnb    = (const float*)d_in[16];
    const float* Wq     = (const float*)d_in[17];
    const float* bq     = (const float*)d_in[18];
    const float* Wk     = (const float*)d_in[19];
    const float* bk     = (const float*)d_in[20];
    const float* Wv     = (const float*)d_in[21];
    const float* bvp    = (const float*)d_in[22];
    const float* Wo     = (const float*)d_in[23];
    const float* bo     = (const float*)d_in[24];
    const float* ln1g   = (const float*)d_in[25];
    const float* ln1b   = (const float*)d_in[26];
    const float* W1     = (const float*)d_in[27];
    const float* b1     = (const float*)d_in[28];
    const float* W2     = (const float*)d_in[29];
    const float* b2     = (const float*)d_in[30];
    const float* ln2g   = (const float*)d_in[31];
    const float* ln2b   = (const float*)d_in[32];
    (void)in_sizes; (void)n_in; (void)out_size; (void)ws_size;

    const long XSZ = (long)NBATCH * LTOT * DMODEL;    // 5,570,560
    const long XBS = (long)LTOT * DMODEL;             // per-batch stride
    const int  MROWS = NBATCH * LTOT;                 // 10880
    const long VTB = (long)NHEAD * DKV * LTOT;        // 348,160 per batch
    const long SCZ = (long)NHEAD * LTOT * LTOT;       // 3,699,200

    // f32 region
    float* ws  = (float*)d_ws;
    float* X   = ws;
    float* TMP = X   + XSZ;
    float* H1  = TMP + XSZ;                           // 10880 x 2048
    float* SC  = H1  + (long)MROWS * DFF;             // per-batch f32 scores
    float* C1  = SC  + SCZ;
    float* C2  = C1  + (long)NBATCH * 128 * DBN;
    float* C3  = C2  + (long)NBATCH * 32 * DBN;
    float* ST  = C3  + (long)NBATCH * 8 * DBN;        // 256 + pad
    // f16 region
    _Float16* Xh  = (_Float16*)(ST + 1024);
    _Float16* Qh  = Xh  + XSZ;
    _Float16* Kh  = Qh  + XSZ;
    _Float16* VT  = Kh  + XSZ;                        // [b][h][64][680]
    _Float16* Oh  = VT  + XSZ;
    _Float16* SCh = Oh  + XSZ;                        // per-batch f16 probs
    _Float16* H1h = SCh + SCZ;
    _Float16* T0h = H1h + (long)MROWS * DFF;
    _Float16* C1h = T0h + (long)NBATCH * SEQLEN * DBN;
    _Float16* C2h = C1h + (long)NBATCH * 128 * DBN;
    _Float16* C3h = C2h + (long)NBATCH * 32 * DBN;
    _Float16* Wt  = C3h + (long)NBATCH * 8 * DBN;
    // transposed f16 weight pool
    _Float16* downWt = Wt;                            // [128][512]
    _Float16* convKt = downWt + 65536;                // 3x[128][512]
    _Float16* upWt   = convKt + 196608;               // [512][128]
    _Float16* Wqt    = upWt   + 65536;                // 4x[512][512]
    _Float16* Wkt    = Wqt    + 1048576;
    _Float16* Wvt    = Wkt    + 1048576;
    _Float16* Wot    = Wvt    + 1048576;
    _Float16* W1t    = Wot    + 1048576;              // 4x[2048][512]
    _Float16* W2t    = W1t    + 4194304;              // 4x[512][2048]
    unsigned long long* rmask = (unsigned long long*)(W2t + 4194304); // [43]

    // 0) one-time weight transposes + mask tile bitmask
    auto tr = [&](const float* in, _Float16* out, int K, int N) {
        long n = (long)K * N;
        wtrans_kernel<<<(unsigned)((n + 255) / 256), 256, 0, stream>>>(in, out, K, N);
    };
    tr(downW, downWt, 512, 128);
    for (int i = 0; i < 3; ++i) tr(convK + (long)i * 65536, convKt + (long)i * 65536, 512, 128);
    tr(upW, upWt, 128, 512);
    for (int l = 0; l < NLAYER; ++l) {
        tr(Wq + (long)l * 262144, Wqt + (long)l * 262144, 512, 512);
        tr(Wk + (long)l * 262144, Wkt + (long)l * 262144, 512, 512);
        tr(Wv + (long)l * 262144, Wvt + (long)l * 262144, 512, 512);
        tr(Wo + (long)l * 262144, Wot + (long)l * 262144, 512, 512);
        tr(W1 + (long)l * 1048576, W1t + (long)l * 1048576, 512, 2048);
        tr(W2 + (long)l * 1048576, W2t + (long)l * 1048576, 2048, 512);
    }
    tilemap_kernel<<<1, 64, 0, stream>>>(mask, rmask);

    // 1) embedding -> X,Xh rows [0,512)
    embed_kernel<<<(NBATCH * SEQLEN * DMODEL) / 256, 256, 0, stream>>>(
        x_enc, x_mark, tokK, markW, markb, X, Xh);

    // 2) down projection -> T0h f16 [b,512,128]
    launch_gemm(G_F16, Xh, downWt, T0h, downb, SEQLEN, DBN, DMODEL,
                DMODEL, DMODEL, DBN, XBS, 0, 0, 0, (long)SEQLEN * DBN, 0,
                NBATCH, 1, nullptr, stream);

    // 3) conv pyramid (stride-4 k=4 == reshape+GEMM), BN stats f32, ELU->f16
    struct { _Float16* inh; float* outf; _Float16* outh; int Mrows; } lv[3] = {
        { T0h, C1, C1h, NBATCH * 128 },
        { C1h, C2, C2h, NBATCH * 32  },
        { C2h, C3, C3h, NBATCH * 8   },
    };
    for (int i = 0; i < 3; ++i) {
        launch_gemm(G_F32, lv[i].inh, convKt + (long)i * 65536, lv[i].outf,
                    convb + i * DBN, lv[i].Mrows, DBN, 4 * DBN,
                    4 * DBN, 4 * DBN, DBN, 0, 0, 0, 0, 0, 0,
                    1, 1, nullptr, stream);
        bn_stats_kernel<<<DBN, 256, 0, stream>>>(lv[i].outf, lv[i].Mrows, ST);
        long n = (long)lv[i].Mrows * DBN;
        bn_elu_kernel<<<(unsigned)((n + 255) / 256), 256, 0, stream>>>(
            lv[i].outf, lv[i].outh, ST, bng + i * DBN, bnb + i * DBN, n);
    }

    // 4) up projection into X rows [512,680)
    launch_gemm(G_F32, C1h, upWt, X + (long)512 * DMODEL, upb, 128, DMODEL, DBN,
                DBN, DBN, DMODEL, (long)128 * DBN, 0, 0, 0, XBS, 0,
                NBATCH, 1, nullptr, stream);
    launch_gemm(G_F32, C2h, upWt, X + (long)640 * DMODEL, upb, 32, DMODEL, DBN,
                DBN, DBN, DMODEL, (long)32 * DBN, 0, 0, 0, XBS, 0,
                NBATCH, 1, nullptr, stream);
    launch_gemm(G_F32, C3h, upWt, X + (long)672 * DMODEL, upb, 8, DMODEL, DBN,
                DBN, DBN, DMODEL, (long)8 * DBN, 0, 0, 0, XBS, 0,
                NBATCH, 1, nullptr, stream);

    // 5) concat-LayerNorm
    ln_kernel<<<MROWS, 256, 0, stream>>>(X, nullptr, X, Xh, cng, cnb);

    // 6) transformer layers
    for (int l = 0; l < NLAYER; ++l) {
        const long WS2 = (long)l * 262144;
        // Q,K -> f16 [10880,512]
        launch_gemm(G_F16, Xh, Wqt + WS2, Qh, bq + l * DMODEL, MROWS, DMODEL, DMODEL,
                    DMODEL, DMODEL, DMODEL, 0, 0, 0, 0, 0, 0, 1, 1, nullptr, stream);
        launch_gemm(G_F16, Xh, Wkt + WS2, Kh, bk + l * DMODEL, MROWS, DMODEL, DMODEL,
                    DMODEL, DMODEL, DMODEL, 0, 0, 0, 0, 0, 0, 1, 1, nullptr, stream);
        // V stored transposed: VT[b][h][d][l]
        launch_gemm(G_F16T, Xh, Wvt + WS2, VT, bvp + l * DMODEL, LTOT, DMODEL, DMODEL,
                    DMODEL, DMODEL, LTOT, XBS, 0, 0, 0, VTB, 0,
                    NBATCH, 1, nullptr, stream);

        for (int b = 0; b < NBATCH; ++b) {
            // scores[h,q,k] = q.k ; fully-masked tile groups skipped
            launch_gemm(G_F32_MN, Qh + (long)b * XBS, Kh + (long)b * XBS, SC, nullptr,
                        LTOT, LTOT, DKV, DMODEL, DMODEL, LTOT,
                        0, DKV, 0, DKV, 0, (long)LTOT * LTOT,
                        1, NHEAD, rmask, stream);
            softmax_kernel<<<dim3(LTOT, NHEAD), 256, 0, stream>>>(SC, SCh, mask);
            // o[q, h*64+d] = probs @ v ; zero-prob K-blocks skipped
            launch_gemm(G_F16_MK, SCh, VT + (long)b * VTB, Oh + (long)b * XBS, nullptr,
                        LTOT, DKV, LTOT, LTOT, LTOT, DMODEL,
                        0, (long)LTOT * LTOT, 0, (long)DKV * LTOT, 0, DKV,
                        1, NHEAD, rmask, stream);
        }

        launch_gemm(G_F32, Oh, Wot + WS2, TMP, bo + l * DMODEL, MROWS, DMODEL, DMODEL,
                    DMODEL, DMODEL, DMODEL, 0, 0, 0, 0, 0, 0, 1, 1, nullptr, stream);
        ln_kernel<<<MROWS, 256, 0, stream>>>(TMP, X, X, Xh,
                                             ln1g + l * DMODEL, ln1b + l * DMODEL);

        launch_gemm(G_F32, Xh, W1t + (long)l * 1048576, H1, b1 + l * DFF,
                    MROWS, DFF, DMODEL, DMODEL, DMODEL, DFF,
                    0, 0, 0, 0, 0, 0, 1, 1, nullptr, stream);
        long nh = (long)MROWS * DFF;
        gelu_kernel<<<(unsigned)((nh + 255) / 256), 256, 0, stream>>>(H1, H1h, nh);
        launch_gemm(G_F32, H1h, W2t + (long)l * 1048576, TMP, b2 + l * DMODEL,
                    MROWS, DMODEL, DFF, DFF, DFF, DMODEL,
                    0, 0, 0, 0, 0, 0, 1, 1, nullptr, stream);
        ln_kernel<<<MROWS, 256, 0, stream>>>(TMP, X, X, Xh,
                                             ln2g + l * DMODEL, ln2b + l * DMODEL);
    }

    // 7) gather -> output [16,2048,512] (flat == [16,512,2048])
    long nout = (long)NBATCH * 2048 * DMODEL;
    gather_kernel<<<(unsigned)((nout + 255) / 256), 256, 0, stream>>>(
        X, gidx, (float*)d_out);
}
